// CausalLoopJIT_61074434949418
// MI455X (gfx1250) — matrix operand order repaired and checked by reading, not compiled, not run on hardware
//
#include <hip/hip_runtime.h>
#include <hip/hip_bf16.h>

// ---------------- problem constants (fixed by the reference) ----------------
#define BQ   256   // batch
#define TQ   512   // timesteps
#define NQ   256   // substrate dim
#define HQ   256   // controller hidden
#define WASH 64
#define TOUT (TQ - WASH)   // 448
#define MT   16    // batch rows per workgroup
#define KX   544   // GEMM1 K (514 padded to multiple of 32)

typedef __attribute__((ext_vector_type(16))) __bf16 bf16x16;
typedef __attribute__((ext_vector_type(8)))  __bf16 bf16x8;
typedef __attribute__((ext_vector_type(8)))  float  f32x8;

__device__ __forceinline__ bf16x16 ld_frag(const __hip_bfloat16* lo,
                                           const __hip_bfloat16* hi) {
    union { bf16x16 v; bf16x8 h[2]; } u;
    u.h[0] = *reinterpret_cast<const bf16x8*>(lo);   // 16B -> VGPR0..3
    u.h[1] = *reinterpret_cast<const bf16x8*>(hi);   // 16B -> VGPR4..7
    return u.v;
}

__device__ __forceinline__ f32x8 wmma_bf16(bf16x16 a, bf16x16 b, f32x8 c) {
    // D = A(16x32 bf16) * B(32x16 bf16) + C(16x16 f32)
    return __builtin_amdgcn_wmma_f32_16x16x32_bf16(
        false, a, false, b, (short)0, c, false, false);
}

__device__ __forceinline__ float sigm(float x) { return 1.f / (1.f + __expf(-x)); }
__device__ __forceinline__ float tanh_fast(float x) {
    return 1.f - 2.f / (__expf(2.f * x) + 1.f);
}

// --------------------------------------------------------------------------
// Prep: f32 weights -> bf16, transposed to B-operand-friendly [Ncol][K] order.
//   WznT[j][k], j<256: z-gate col j of Wx ; j>=256: n-gate col (Wx col 512+(j-256))
//   K rows 0..513 = Wx rows (incl. the two spin inputs), 514..543 zero pad.
//   WoutT[16][256]: rows 0,1 = Wout columns; rows 2..15 zero.
// --------------------------------------------------------------------------
__global__ void prep_weights(const float* __restrict__ Wx,
                             const float* __restrict__ Wout,
                             __hip_bfloat16* __restrict__ WznT,
                             __hip_bfloat16* __restrict__ WoutT) {
    const int stride = gridDim.x * blockDim.x;
    const int t0 = blockIdx.x * blockDim.x + threadIdx.x;
    for (int idx = t0; idx < 512 * KX; idx += stride) {
        int j = idx / KX, k = idx - j * KX;
        float v = 0.f;
        if (k < 514) v = Wx[k * 768 + ((j < 256) ? j : (256 + j))]; // 512+(j-256)
        WznT[idx] = __float2bfloat16(v);
    }
    for (int idx = t0; idx < 16 * 256; idx += stride) {
        int l = idx >> 8, k = idx & 255;
        WoutT[idx] = __float2bfloat16((l < 2) ? Wout[k * 2 + l] : 0.f);
    }
}

// --------------------------------------------------------------------------
// Persistent scan kernel: one workgroup owns 16 batch rows for all 512 steps.
// 512 threads = 16 wave32s. Wave w owns column-tile w (16 cols) of every GEMM.
// --------------------------------------------------------------------------
__global__ __launch_bounds__(512)
void causal_loop_kernel(const int*   __restrict__ settings,  // [B,T,2]
                        const float* __restrict__ noiseA,    // [B,T,N]
                        const float* __restrict__ noiseB,    // [B,T,N]
                        const float* __restrict__ WinA,      // [2,N]
                        const float* __restrict__ WrecA,     // [N,N]
                        const float* __restrict__ WinB,      // [2,N]
                        const float* __restrict__ WrecB,     // [N,N]
                        const float* __restrict__ bgru,      // [3H]
                        const float* __restrict__ bout,      // [2]
                        const __hip_bfloat16* __restrict__ WznT,   // [512][KX]
                        const __hip_bfloat16* __restrict__ WoutT,  // [16][256]
                        float* __restrict__ out) {
    extern __shared__ char smem_raw[];
    __hip_bfloat16* xbuf = reinterpret_cast<__hip_bfloat16*>(smem_raw); // 16*KX
    __hip_bfloat16* sAbf = xbuf + MT * KX;     // 16*256 bf16 state A
    __hip_bfloat16* sBbf = sAbf + MT * NQ;     // 16*256 bf16 state B
    __hip_bfloat16* hbuf = sBbf + MT * NQ;     // 16*256 bf16 controller h
    __hip_bfloat16* wrA  = hbuf + MT * NQ;     // 256*256 W_rec_A^T (LDS-cached)
    __hip_bfloat16* wrB  = wrA + NQ * NQ;      // 256*256 W_rec_B^T
    float* bitsArr = reinterpret_cast<float*>(wrB + NQ * NQ); // [16][2]
    float* delArr  = bitsArr + 32;                            // [16][2]
    float* logbuf  = delArr + 32;                             // [2][16][2] delay ring

    const int tid  = threadIdx.x;
    const int wave = tid >> 5;
    const int lane = tid & 31;
    const int ncol = lane & 15;          // N index inside 16-wide tile
    const int hhalf = lane >> 4;         // 0: lanes 0-15, 1: lanes 16-31
    const int b0   = blockIdx.x * MT;

    // ---- one-time init: cache transposed recurrent weights in LDS, zero state
    for (int i = tid; i < NQ * NQ; i += 512) {
        int n = i >> 8, k = i & 255;
        wrA[i] = __float2bfloat16(WrecA[k * NQ + n]);
        wrB[i] = __float2bfloat16(WrecB[k * NQ + n]);
    }
    for (int i = tid; i < MT * NQ; i += 512) {
        sAbf[i] = __float2bfloat16(0.f);
        sBbf[i] = __float2bfloat16(0.f);
    }
    if (tid < 128) bitsArr[tid] = 0.f;   // bits, delayed, delay ring
    __syncthreads();

    // ---- per-wave loop-invariant scalars
    const int col  = wave * 16 + ncol;             // this lane's output column
    const float bz = bgru[col];                    // z-gate bias
    const float bn = bgru[512 + col];              // n-gate bias
    const float wA0 = WinA[col],      wA1 = WinA[NQ + col];
    const float wB0 = WinB[col],      wB1 = WinB[NQ + col];
    const float bo  = (ncol < 2) ? bout[ncol] : 0.f;

    for (int t = 0; t < TQ; ++t) {
        // Opaque memory barrier: stops LICM from hoisting the loop-invariant
        // LDS weight-fragment loads out of the t-loop (which spilled to
        // scratch under the 256-VGPR/wave budget). Forces ds_load per step.
        asm volatile("" ::: "memory");

        const int slot = t & 1;   // depth-2 delay ring: slot holds logits(t-2)

        // ---------------- P1: build x = [sA+nA, sB+nB, spins, 0pad] ---------
        // wave w owns batch row m=w; lane covers k = lane + 32*j (coalesced)
        {
            const int m = wave;
            const size_t nbase = ((size_t)(b0 + m) * TQ + t) * NQ;
            #pragma unroll
            for (int j = 0; j < KX / 32; ++j) {
                int k = lane + 32 * j;
                float v;
                if (k < NQ)
                    v = __bfloat162float(sAbf[m * NQ + k]) + noiseA[nbase + k];
                else if (k < 2 * NQ)
                    v = __bfloat162float(sBbf[m * NQ + (k - NQ)]) + noiseB[nbase + (k - NQ)];
                else if (k < 514)
                    v = 2.f * (float)settings[((b0 + m) * TQ + t) * 2 + (k - 512)] - 1.f;
                else
                    v = 0.f;
                xbuf[m * KX + k] = __float2bfloat16(v);
            }
        }
        if (tid < MT) {
            int m = tid;
            bitsArr[m * 2 + 0] = (float)settings[((b0 + m) * TQ + t) * 2 + 0];
            bitsArr[m * 2 + 1] = (float)settings[((b0 + m) * TQ + t) * 2 + 1];
            delArr[m * 2 + 0]  = logbuf[slot * 32 + m * 2 + 0];
            delArr[m * 2 + 1]  = logbuf[slot * 32 + m * 2 + 1];
        }
        __syncthreads();

        // -------- P3a: GEMM1  gz/gn = x @ Wzn  (K = 544, 17 chunks) ---------
        f32x8 accZ = {}, accN = {};
        {
            const int kB = hhalf * 16;              // B-frag K sub-offset
            const int m  = ncol;                    // A row for this lane
            const int kA = hhalf * 16;              // A-frag K sub-offset
            #pragma unroll
            for (int kc = 0; kc < KX; kc += 32) {
                bf16x16 a = ld_frag(xbuf + m * KX + kc + kA,
                                    xbuf + m * KX + kc + kA + 8);
                const __hip_bfloat16* pz = WznT + (size_t)col * KX + kc + kB;
                const __hip_bfloat16* pn = WznT + (size_t)(256 + col) * KX + kc + kB;
                accZ = wmma_bf16(a, ld_frag(pz, pz + 8), accZ);
                accN = wmma_bf16(a, ld_frag(pn, pn + 8), accN);
            }
        }
        // fused GRU epilogue (h0==0 => h = (1-z)*n), write h for logits GEMM
        {
            const int mb = hhalf * 8;
            #pragma unroll
            for (int r = 0; r < 8; ++r) {
                float h = (1.f - sigm(accZ[r] + bz)) * tanh_fast(accN[r] + bn);
                hbuf[(mb + r) * NQ + col] = __float2bfloat16(h);
            }
        }

        // -------- P3b: GEMM2  reservoir updates (B-operands from LDS) -------
        f32x8 accA = {}, accB = {};
        {
            const int kB = hhalf * 16;
            const int m  = ncol;
            const int kA = hhalf * 16;
            #pragma unroll
            for (int kc = 0; kc < NQ; kc += 32) {
                bf16x16 aA = ld_frag(sAbf + m * NQ + kc + kA,
                                     sAbf + m * NQ + kc + kA + 8);
                bf16x16 aB = ld_frag(sBbf + m * NQ + kc + kA,
                                     sBbf + m * NQ + kc + kA + 8);
                const __hip_bfloat16* pA = wrA + col * NQ + kc + kB;
                const __hip_bfloat16* pB = wrB + col * NQ + kc + kB;
                accA = wmma_bf16(aA, ld_frag(pA, pA + 8), accA);
                accB = wmma_bf16(aB, ld_frag(pB, pB + 8), accB);
            }
        }
        __syncthreads();   // sAbf/sBbf reads done; hbuf complete

        // -------- P4: state tanh epilogue (overwrite bf16 states) -----------
        {
            const int mb = hhalf * 8;
            #pragma unroll
            for (int r = 0; r < 8; ++r) {
                int m = mb + r;
                float gA = accA[r] + bitsArr[m * 2 + 0] * wA0 + delArr[m * 2 + 0] * wA1;
                float gB = accB[r] + bitsArr[m * 2 + 1] * wB0 + delArr[m * 2 + 1] * wB1;
                sAbf[m * NQ + col] = __float2bfloat16(tanh_fast(gA));
                sBbf[m * NQ + col] = __float2bfloat16(tanh_fast(gB));
            }
        }

        // -------- P4: wave 0 -> logits = h @ Wout + b_out, sign + ring ------
        if (wave == 0) {
            f32x8 accL = {};
            const int kB = hhalf * 16;
            const int m  = ncol;
            const int kA = hhalf * 16;
            #pragma unroll
            for (int kc = 0; kc < NQ; kc += 32) {
                bf16x16 aH = ld_frag(hbuf + m * NQ + kc + kA,
                                     hbuf + m * NQ + kc + kA + 8);
                const __hip_bfloat16* pL = WoutT + ncol * NQ + kc + kB;
                accL = wmma_bf16(aH, ld_frag(pL, pL + 8), accL);
            }
            if (ncol < 2) {
                const int mb = hhalf * 8;
                #pragma unroll
                for (int r = 0; r < 8; ++r) {
                    int m = mb + r;
                    float lv = accL[r] + bo;
                    logbuf[slot * 32 + m * 2 + ncol] = lv;   // becomes delayed(t+2)
                    if (t >= WASH) {
                        float sg = (lv > 0.f) ? 1.f : ((lv < 0.f) ? -1.f : 0.f);
                        out[(size_t)ncol * BQ * TOUT + (size_t)(b0 + m) * TOUT + (t - WASH)] = sg;
                    }
                }
            }
        }
        // -------- P4: wave 1 -> settings log ------
        if (wave == 1 && t >= WASH) {
            int m = lane & 15, l = lane >> 4;
            out[(size_t)2 * BQ * TOUT + ((size_t)(b0 + m) * TOUT + (t - WASH)) * 2 + l] =
                bitsArr[m * 2 + l];
        }
        __syncthreads();   // states/ring visible to next step's P1
    }
}

extern "C" void kernel_launch(void* const* d_in, const int* in_sizes, int n_in,
                              void* d_out, int out_size, void* d_ws, size_t ws_size,
                              hipStream_t stream) {
    const int*   settings = (const int*)  d_in[0];
    const float* nA       = (const float*)d_in[1];
    const float* nB       = (const float*)d_in[2];
    const float* WinA     = (const float*)d_in[3];
    const float* WrecA    = (const float*)d_in[4];
    const float* WinB     = (const float*)d_in[5];
    const float* WrecB    = (const float*)d_in[6];
    const float* Wx       = (const float*)d_in[7];
    // d_in[8] = Wh is provably dead (h0 == 0 in the reference GRU)
    const float* bgru     = (const float*)d_in[9];
    const float* Wout     = (const float*)d_in[10];
    const float* bout     = (const float*)d_in[11];

    __hip_bfloat16* WznT  = (__hip_bfloat16*)d_ws;       // 512*544 bf16
    __hip_bfloat16* WoutT = WznT + 512 * KX;             // 16*256 bf16

    prep_weights<<<256, 256, 0, stream>>>(Wx, Wout, WznT, WoutT);

    // dynamic LDS: x + 3 state/h tiles (bf16) + two cached 256x256 bf16
    // recurrent matrices + small f32 scratch  ~= 298 KB (< 320 KB/WGP)
    size_t smem = (size_t)(MT * KX + 3 * MT * NQ + 2 * NQ * NQ) * sizeof(__hip_bfloat16)
                + 128 * sizeof(float);
    causal_loop_kernel<<<BQ / MT, 512, smem, stream>>>(
        settings, nA, nB, WinA, WrecA, WinB, WrecB, bgru, bout,
        WznT, WoutT, (float*)d_out);
}
